// STMetaLSTM_17995912970448
// MI455X (gfx1250) — matrix operand order repaired and checked by reading, not compiled, not run on hardware
//
#include <hip/hip_runtime.h>

#define BB   32
#define TT   12
#define NNODE 325
#define HH   64
#define EE   96
#define LHH  128
#define OUTT 12
#define BN   (BB * NNODE)   // 10400

typedef __attribute__((ext_vector_type(16))) _Float16 v16h;
typedef __attribute__((ext_vector_type(8)))  _Float16 v8h;
typedef __attribute__((ext_vector_type(8)))  float    v8f;

// ---------------------------------------------------------------- utilities
__global__ void f32_to_f16_kernel(const float* __restrict__ src,
                                  _Float16* __restrict__ dst, long count) {
  long i = (long)blockIdx.x * blockDim.x + threadIdx.x;
  if (i < count) dst[i] = (_Float16)src[i];
}

// meta[b,n,e] = mean_t x[b,t,n,1+e]   -> f16
__global__ void meta_kernel(const float* __restrict__ x,
                            _Float16* __restrict__ meta) {
  int idx = blockIdx.x * blockDim.x + threadIdx.x;
  if (idx >= BN * EE) return;
  int e  = idx % EE;
  int bn = idx / EE;
  int n  = bn % NNODE, b = bn / NNODE;
  float s = 0.f;
#pragma unroll
  for (int t = 0; t < TT; ++t)
    s += x[(((long)b * TT + t) * NNODE + n) * (EE + 1) + 1 + e];
  meta[(long)bn * EE + e] = (_Float16)(s * (1.0f / TT));
}

// ---------------------------------------------------------------- WMMA GEMM
// Batched C = act(A(f16, MxK) * B(f16, KxN) + bias(f32, N))
// Register-tiled: each wave computes a (16*TM) x (16*TN) block of C, reusing
// TM A-fragments against TN B-fragments -> TM*TN WMMAs per K-step.
// Fragment layouts per CDNA5 ISA (CDNA5 ISA):
//   A (16x32 f16): lane m = lane&15; halves 0..7  = K (base..base+7),
//                  halves 8..15 = K (16+base..16+base+7), base = (lane<16)?0:8
//   B (32x16 f16): lane n = lane&15; halves 0..15 = K in the same order as A
//   C (16x16 f32): lane<16: n=lane, rows m0..m0+7 ; lane>=16: n=lane-16, rows m0+8..m0+15
template <int TM, int TN>
__global__ void __launch_bounds__(256)
gemm_wmma_kernel(const _Float16* __restrict__ A,  long sAb, int lda,
                 const _Float16* __restrict__ Bm, long sBb, int ldb,
                 const float*    __restrict__ bias, long sBiasb,
                 void* __restrict__ Cout, long sCb, int ldc,
                 int M, int N, int K, int relu, int out_f16)
{
  const int lane = threadIdx.x & 31;
  const int wave = threadIdx.x >> 5;
  const int n0   = (blockIdx.x * 8 + wave) * (16 * TN);
  const int m0   = blockIdx.y * (16 * TM);
  const int batch = blockIdx.z;
  if (n0 >= N || m0 >= M) return;   // wave-uniform; EXEC stays all-ones

  const _Float16* Ab  = A  + (long)batch * sAb;
  const _Float16* Bb_ = Bm + (long)batch * sBb;
  const float*    bi  = bias + (long)batch * sBiasb;

  const int mrow = lane & 15;
  const int kA   = (lane < 16) ? 0 : 8;

  v8f acc[TM][TN];
#pragma unroll
  for (int mi = 0; mi < TM; ++mi)
#pragma unroll
    for (int ni = 0; ni < TN; ++ni) acc[mi][ni] = (v8f){};

  for (int k0 = 0; k0 < K; k0 += 32) {
    v16h afr[TM];
#pragma unroll
    for (int mi = 0; mi < TM; ++mi) {
      const _Float16* ap = Ab + (long)(m0 + mi * 16 + mrow) * lda + k0 + kA;
      if (TM > 1 && k0 + 32 < K)
        __builtin_prefetch(ap + 32, 0, 3);     // -> global_prefetch_b8
      v8h alo = *(const v8h*)(ap);
      v8h ahi = *(const v8h*)(ap + 16);
      afr[mi] = __builtin_shufflevector(alo, ahi,
          0,1,2,3,4,5,6,7,8,9,10,11,12,13,14,15);
    }
    v16h bfr[TN];
    const _Float16* bprow = Bb_ + (long)(k0 + kA) * ldb + n0 + (lane & 15);
#pragma unroll
    for (int ni = 0; ni < TN; ++ni) {
#pragma unroll
      for (int s = 0; s < 16; ++s)
        bfr[ni][s] = bprow[(long)(s < 8 ? s : s + 8) * ldb + ni * 16];
    }
#pragma unroll
    for (int mi = 0; mi < TM; ++mi)
#pragma unroll
      for (int ni = 0; ni < TN; ++ni)
        acc[mi][ni] = __builtin_amdgcn_wmma_f32_16x16x32_f16(
            /*neg_a=*/false, afr[mi], /*neg_b=*/false, bfr[ni],
            /*c_mod=*/(short)0, acc[mi][ni],
            /*reuse_a=*/false, /*reuse_b=*/false);
  }

#pragma unroll
  for (int mi = 0; mi < TM; ++mi) {
    const int mbase = m0 + mi * 16 + ((lane < 16) ? 0 : 8);
#pragma unroll
    for (int ni = 0; ni < TN; ++ni) {
      const int ncol = n0 + ni * 16 + (lane & 15);
      const float bval = bi[ncol];
#pragma unroll
      for (int r = 0; r < 8; ++r) {
        float v = acc[mi][ni][r] + bval;
        if (relu) v = v > 0.f ? v : 0.f;
        long idx = (long)batch * sCb + (long)(mbase + r) * ldc + ncol;
        if (out_f16) ((_Float16*)Cout)[idx] = (_Float16)v;
        else         ((float*)Cout)[idx]    = v;
      }
    }
  }
}

// ---------------------------------------------------------------- fused LSTM
// One workgroup per (b,n) cell. 256 threads: tid -> (gate g = tid>>6, k = tid&63).
// Each thread keeps its Wh column Wh[g][:,k] (64 f32) in registers for all 12
// steps; h/c/z exchanged through LDS. Then the ReLU->32->ReLU->12 head.
__global__ void __launch_bounds__(256)
lstm_head_kernel(const float* __restrict__ x,       // (B,T,N,97)
                 const _Float16* __restrict__ Wh,   // (4,BN,4096)
                 const float* __restrict__ Wx,      // (4,BN,64)
                 const float* __restrict__ bbv_,    // (4,BN,64)
                 const float* __restrict__ fc1, const float* __restrict__ fc1b,
                 const float* __restrict__ fc2, const float* __restrict__ fc2b,
                 float* __restrict__ out)           // (B,12,N)
{
  __shared__ float hbuf[HH], cbuf[HH], zbuf[256], a1[HH / 2];

  const int bn = blockIdx.x;
  const int n = bn % NNODE, b = bn / NNODE;
  const int tid = threadIdx.x;
  const int g = tid >> 6, k = tid & 63;

  // per-thread column of Wh (coalesced: consecutive k -> consecutive halves)
  float wcol[HH];
  const _Float16* whp = Wh + ((long)g * BN + bn) * (HH * HH) + k;
#pragma unroll
  for (int h = 0; h < HH; ++h) wcol[h] = (float)whp[h * HH];

  const float wxv = Wx[((long)g * BN + bn) * HH + k];
  const float bv  = bbv_[((long)g * BN + bn) * HH + k];

  if (tid < HH) { hbuf[tid] = 0.f; cbuf[tid] = 0.f; }
  __syncthreads();

  for (int t = 0; t < TT; ++t) {
    const float xt = x[(((long)b * TT + t) * NNODE + n) * (EE + 1)];
    float z = xt * wxv + bv;
#pragma unroll
    for (int h = 0; h < HH; ++h) z += wcol[h] * hbuf[h];
    zbuf[tid] = z;
    __syncthreads();
    if (tid < HH) {
      float gg = tanhf(zbuf[tid]);
      float ii = 1.f / (1.f + expf(-zbuf[64  + tid]));
      float ff = 1.f / (1.f + expf(-zbuf[128 + tid]));
      float oo = 1.f / (1.f + expf(-zbuf[192 + tid]));
      float cc = gg * ii + cbuf[tid] * ff;
      cbuf[tid] = cc;
      hbuf[tid] = tanhf(cc) * oo;
    }
    __syncthreads();
  }

  // head: relu(h) @ fc1(64x32) + b -> relu -> @ fc2(32x12) + b
  if (tid < HH) hbuf[tid] = fmaxf(hbuf[tid], 0.f);
  __syncthreads();
  if (tid < HH / 2) {
    float s = fc1b[tid];
#pragma unroll
    for (int kk = 0; kk < HH; ++kk) s += hbuf[kk] * fc1[kk * (HH / 2) + tid];
    a1[tid] = fmaxf(s, 0.f);
  }
  __syncthreads();
  if (tid < OUTT) {
    float s = fc2b[tid];
#pragma unroll
    for (int j = 0; j < HH / 2; ++j) s += a1[j] * fc2[j * OUTT + tid];
    out[((long)b * OUTT + tid) * NNODE + n] = s;
  }
}

// ---------------------------------------------------------------- launch
static inline size_t alignUp(size_t v) { return (v + 255) & ~(size_t)255; }

extern "C" void kernel_launch(void* const* d_in, const int* in_sizes, int n_in,
                              void* d_out, int out_size, void* d_ws, size_t ws_size,
                              hipStream_t stream) {
  (void)in_sizes; (void)n_in; (void)out_size; (void)ws_size;
  const float* x    = (const float*)d_in[0];
  const float* wx1  = (const float*)d_in[1];
  const float* wx1b = (const float*)d_in[2];
  const float* wx2  = (const float*)d_in[3];
  const float* wx2b = (const float*)d_in[4];
  const float* wh1  = (const float*)d_in[5];
  const float* wh1b = (const float*)d_in[6];
  const float* wh2  = (const float*)d_in[7];
  const float* wh2b = (const float*)d_in[8];
  const float* b1   = (const float*)d_in[9];
  const float* b1b  = (const float*)d_in[10];
  const float* b2   = (const float*)d_in[11];
  const float* b2b  = (const float*)d_in[12];
  const float* fc1  = (const float*)d_in[13];
  const float* fc1b = (const float*)d_in[14];
  const float* fc2  = (const float*)d_in[15];
  const float* fc2b = (const float*)d_in[16];
  float* out = (float*)d_out;

  // workspace layout
  char* ws = (char*)d_ws;
  size_t off = 0;
  auto take = [&](size_t bytes) { size_t o = off; off = alignUp(off + bytes); return o; };
  _Float16* meta_h  = (_Float16*)(ws + take((size_t)BN * EE * 2));
  _Float16* wx1_h   = (_Float16*)(ws + take((size_t)4 * EE * LHH * 2));
  _Float16* wh1_h   = (_Float16*)(ws + take((size_t)4 * EE * LHH * 2));
  _Float16* b1_h    = (_Float16*)(ws + take((size_t)4 * EE * LHH * 2));
  _Float16* wx2_h   = (_Float16*)(ws + take((size_t)4 * LHH * HH * 2));
  _Float16* b2_h    = (_Float16*)(ws + take((size_t)4 * LHH * HH * 2));
  _Float16* wh2_h   = (_Float16*)(ws + take((size_t)4 * LHH * (HH * HH) * 2));
  _Float16* hid     = (_Float16*)(ws + take((size_t)3 * 4 * BN * LHH * 2)); // [hyp][g][BN][LH]
  float*    Wx_f    = (float*)   (ws + take((size_t)4 * BN * HH * 4));
  float*    bb_f    = (float*)   (ws + take((size_t)4 * BN * HH * 4));
  _Float16* Wh_h    = (_Float16*)(ws + take((size_t)4 * BN * (HH * HH) * 2));

  // 1) weight conversions f32 -> f16
  auto cvt = [&](const float* s, _Float16* d, long cnt) {
    f32_to_f16_kernel<<<dim3((unsigned)((cnt + 255) / 256)), dim3(256), 0, stream>>>(s, d, cnt);
  };
  cvt(wx1, wx1_h, 4L * EE * LHH);
  cvt(wh1, wh1_h, 4L * EE * LHH);
  cvt(b1,  b1_h,  4L * EE * LHH);
  cvt(wx2, wx2_h, 4L * LHH * HH);
  cvt(b2,  b2_h,  4L * LHH * HH);
  cvt(wh2, wh2_h, 4L * LHH * HH * HH);

  // 2) meta
  meta_kernel<<<dim3((BN * EE + 255) / 256), dim3(256), 0, stream>>>(x, meta_h);

  const int MT = BN / 16; // 650
  // 3) hypernet layer 1 (relu), K=96, N=128, batch 4 per hypernet
  //    hid layout: hyp 0 = wx, 1 = wh, 2 = b
  gemm_wmma_kernel<1, 1><<<dim3(1, MT, 4), dim3(256), 0, stream>>>(
      meta_h, 0, EE, wx1_h, (long)EE * LHH, LHH, wx1b, LHH,
      hid + 0L * 4 * BN * LHH, (long)BN * LHH, LHH, BN, LHH, EE, 1, 1);
  gemm_wmma_kernel<1, 1><<<dim3(1, MT, 4), dim3(256), 0, stream>>>(
      meta_h, 0, EE, wh1_h, (long)EE * LHH, LHH, wh1b, LHH,
      hid + 1L * 4 * BN * LHH, (long)BN * LHH, LHH, BN, LHH, EE, 1, 1);
  gemm_wmma_kernel<1, 1><<<dim3(1, MT, 4), dim3(256), 0, stream>>>(
      meta_h, 0, EE, b1_h, (long)EE * LHH, LHH, b1b, LHH,
      hid + 2L * 4 * BN * LHH, (long)BN * LHH, LHH, BN, LHH, EE, 1, 1);

  // 4) Wh = hid_wh @ wh2 + wh2b  (K=128, N=4096) -> f16
  //    register-tiled: wave = 32x64 block; workgroup = 32x512; grid 8 x 325 x 4
  gemm_wmma_kernel<2, 4><<<dim3(HH * HH / 512, MT / 2, 4), dim3(256), 0, stream>>>(
      hid + 1L * 4 * BN * LHH, (long)BN * LHH, LHH,
      wh2_h, (long)LHH * HH * HH, HH * HH, wh2b, HH * HH,
      Wh_h, (long)BN * HH * HH, HH * HH, BN, HH * HH, LHH, 0, 1);

  // 5) Wx and bias heads (K=128, N=64) -> f32
  gemm_wmma_kernel<1, 1><<<dim3(1, MT, 4), dim3(256), 0, stream>>>(
      hid + 0L * 4 * BN * LHH, (long)BN * LHH, LHH,
      wx2_h, (long)LHH * HH, HH, wx2b, HH,
      Wx_f, (long)BN * HH, HH, BN, HH, LHH, 0, 0);
  gemm_wmma_kernel<1, 1><<<dim3(1, MT, 4), dim3(256), 0, stream>>>(
      hid + 2L * 4 * BN * LHH, (long)BN * LHH, LHH,
      b2_h, (long)LHH * HH, HH, b2b, HH,
      bb_f, (long)BN * HH, HH, BN, HH, LHH, 0, 0);

  // 6) fused LSTM + output head, one workgroup per (b,n)
  lstm_head_kernel<<<dim3(BN), dim3(256), 0, stream>>>(
      x, Wh_h, Wx_f, bb_f, fc1, fc1b, fc2, fc2b, out);
}
